// GATEncoder_18176301596819
// MI455X (gfx1250) — matrix operand order repaired and checked by reading, not compiled, not run on hardware
//
#include <hip/hip_runtime.h>

// ---------------------------------------------------------------------------
// GAT encoder for MI455X (gfx1250, wave32).
// Dense GEMMs: operands pre-packed to bf16 (A row-major K-padded, B transposed)
// so every WMMA fragment is two global_load_b128 -> v_wmma_f32_16x16x32_bf16.
// Segment softmax + scatter -> float atomics with b128 loads.
// Edge-feature projection algebraically folded to a 6xH matrix.
// ---------------------------------------------------------------------------

#define NN    20000
#define EE    320000
#define FIN   55
#define FE    6
#define HIDC  256
#define OUTC  256
#define HH    4
#define ETOT  (EE + NN)
#define NEG_SLOPE 0.2f

typedef __attribute__((ext_vector_type(16))) __bf16 v16bf;
typedef __attribute__((ext_vector_type(8)))  float  v8f;

// ---- helpers ---------------------------------------------------------------

__device__ __forceinline__ float waveReduceSum(float v) {
    #pragma unroll
    for (int off = 16; off > 0; off >>= 1) v += __shfl_xor(v, off, 32);
    return v;
}

// order-preserving float -> uint key (for atomicMax-based segment max)
__device__ __forceinline__ unsigned fkey(float f) {
    unsigned u = __float_as_uint(f);
    return (u & 0x80000000u) ? ~u : (u | 0x80000000u);
}
__device__ __forceinline__ float funkey(unsigned u) {
    return (u & 0x80000000u) ? __uint_as_float(u & 0x7FFFFFFFu)
                             : __uint_as_float(~u);
}

// pack two f32 into a bf16x2 dword (folds to v_cvt_pk_bf16_f32)
__device__ __forceinline__ unsigned pack_bf16x2(float lo, float hi) {
    __bf16 l = (__bf16)lo, h = (__bf16)hi;
    unsigned short lb = __builtin_bit_cast(unsigned short, l);
    unsigned short hb = __builtin_bit_cast(unsigned short, h);
    return ((unsigned)hb << 16) | lb;
}

union fragU { v16bf v; uint4 q[2]; };

// ---- zero fill -------------------------------------------------------------

__global__ void k_zero(float* __restrict__ p, size_t n) {
    for (size_t i = (size_t)blockIdx.x * blockDim.x + threadIdx.x; i < n;
         i += (size_t)gridDim.x * blockDim.x)
        p[i] = 0.0f;
}

// ---- f32 [M,K] -> bf16 [M,Kp] row-major, zero-padded K..Kp -----------------

__global__ void k_cvt_pad(const float* __restrict__ in, unsigned* __restrict__ out,
                          int M, int K, int Kp) {
    size_t idx = (size_t)blockIdx.x * blockDim.x + threadIdx.x;  // over M*Kp/2
    size_t total = (size_t)M * (Kp >> 1);
    if (idx >= total) return;
    int row = (int)(idx / (Kp >> 1));
    int kk  = (int)(idx % (Kp >> 1)) * 2;
    float a0 = (kk     < K) ? in[(size_t)row * K + kk]     : 0.0f;
    float a1 = (kk + 1 < K) ? in[(size_t)row * K + kk + 1] : 0.0f;
    out[idx] = pack_bf16x2(a0, a1);
}

// ---- f32 B[K,Nc] -> bf16 BT[Nc,Kp] (transposed), zero-padded K..Kp ---------

__global__ void k_cvt_T_pad(const float* __restrict__ B, unsigned* __restrict__ out,
                            int K, int Nc, int Kp) {
    size_t idx = (size_t)blockIdx.x * blockDim.x + threadIdx.x;  // over Nc*Kp/2
    size_t total = (size_t)Nc * (Kp >> 1);
    if (idx >= total) return;
    int n  = (int)(idx / (Kp >> 1));
    int kk = (int)(idx % (Kp >> 1)) * 2;
    float b0 = (kk     < K) ? B[(size_t)kk * Nc + n]       : 0.0f;
    float b1 = (kk + 1 < K) ? B[(size_t)(kk + 1) * Nc + n] : 0.0f;
    out[idx] = pack_bf16x2(b0, b1);
}

// ---- column sums of edge_attr (for PyG fill_value='mean' self loops) -------

__global__ void k_ea_colsum(const float* __restrict__ ea, float* __restrict__ sums) {
    float local[FE];
    #pragma unroll
    for (int f = 0; f < FE; ++f) local[f] = 0.0f;
    for (size_t e = (size_t)blockIdx.x * blockDim.x + threadIdx.x; e < EE;
         e += (size_t)gridDim.x * blockDim.x) {
        #pragma unroll
        for (int f = 0; f < FE; ++f) local[f] += ea[e * FE + f];
    }
    #pragma unroll
    for (int f = 0; f < FE; ++f) {
        float v = waveReduceSum(local[f]);
        if ((threadIdx.x & 31) == 0) atomicAdd(&sums[f], v);
    }
}

// ---- fold We (x) a_edge into tiny M matrices: M[f,h] = sum_c We[f,h*C+c]*ae[h,c]

__global__ void k_make_M(const float* __restrict__ We1, const float* __restrict__ ae1,
                         const float* __restrict__ We2, const float* __restrict__ ae2,
                         float* __restrict__ M1, float* __restrict__ M2) {
    int t = threadIdx.x;
    if (t < FE * HH) {
        int f = t / HH, h = t % HH;
        float s = 0.0f;
        for (int c = 0; c < HIDC; ++c)
            s += We1[f * (HH * HIDC) + h * HIDC + c] * ae1[h * HIDC + c];
        M1[f * HH + h] = s;
    } else if (t < FE * HH + FE) {
        int f = t - FE * HH;
        float s = 0.0f;
        for (int c = 0; c < OUTC; ++c)
            s += We2[f * OUTC + c] * ae2[c];
        M2[f] = s;
    }
}

// ---- bf16 WMMA GEMM: C[M,Nc] = A[M,Kp] @ BT[Nc,Kp]^T (pre-packed bf16) -----
// Requirements (guaranteed by launcher): M%16==0, Nc%64==0, Kp%32==0.
// 4 waves / block; 1 wave computes a 16x64 strip (A fragment reused 4x).
// Every fragment is exactly two b128 loads in WMMA register order.

__global__ __launch_bounds__(128) void k_gemm_wmma_bf16(
    const __bf16* __restrict__ A, const __bf16* __restrict__ BT,
    float* __restrict__ C, int M, int Nc, int Kp, int ldc) {
    int groupsN = Nc >> 6;                       // 64-wide column strips
    int wid = blockIdx.x * 4 + (threadIdx.x >> 5);
    if (wid >= (M >> 4) * groupsN) return;       // wave-uniform: EXEC stays all-1s
    int tm = (wid / groupsN) << 4;
    int tn = (wid % groupsN) << 6;

    int lane = threadIdx.x & 31;
    int half = lane >> 4;                        // 0 or 1
    int l    = lane & 15;

    // A fragment (16x32, lane = row tm+l): K runs [k0+8h, +8) and [k0+16+8h, +8)
    const __bf16* arow = A + (size_t)(tm + l) * Kp + half * 8;
    // B fragment (32x16, lane = col tn+16j+l): K runs [k0+8h, +8) and [k0+16+8h, +8)
    const __bf16* brow = BT + (size_t)(tn + l) * Kp + half * 8;

    v8f acc[4] = {{}, {}, {}, {}};
    for (int k0 = 0; k0 < Kp; k0 += 32) {
        fragU af;
        af.q[0] = *(const uint4*)(arow + k0);
        af.q[1] = *(const uint4*)(arow + k0 + 16);
        #pragma unroll
        for (int j = 0; j < 4; ++j) {
            const __bf16* bp = brow + (size_t)j * 16 * Kp + k0;
            fragU bf;
            bf.q[0] = *(const uint4*)(bp);
            bf.q[1] = *(const uint4*)(bp + 16);
            acc[j] = __builtin_amdgcn_wmma_f32_16x16x32_bf16(
                false, af.v, false, bf.v, (short)0, acc[j], false, false);
        }
    }

    // C strip: lane holds (m = r + 8*half, n = tn + 16j + l) in acc[j][r]
    #pragma unroll
    for (int j = 0; j < 4; ++j) {
        #pragma unroll
        for (int r = 0; r < 8; ++r) {
            C[(size_t)(tm + r + half * 8) * ldc + tn + j * 16 + l] = acc[j][r];
        }
    }
}

// ---- per-node attention coefficients: one wave per (node, head) ------------

__global__ __launch_bounds__(256) void k_node_logits(
    const float* __restrict__ h, const float* __restrict__ a_src,
    const float* __restrict__ a_dst, float* __restrict__ alsrc,
    float* __restrict__ aldst, int n, int heads, int C) {
    int wid  = (blockIdx.x * blockDim.x + threadIdx.x) >> 5;
    int lane = threadIdx.x & 31;
    if (wid >= n * heads) return;
    int node = wid / heads, hh = wid % heads;
    const float* row = h + (size_t)node * heads * C + hh * C;
    float s1 = 0.0f, s2 = 0.0f;
    for (int c = lane; c < C; c += 32) {
        float v = row[c];
        s1 += v * a_src[hh * C + c];
        s2 += v * a_dst[hh * C + c];
    }
    s1 = waveReduceSum(s1);
    s2 = waveReduceSum(s2);
    if (lane == 0) { alsrc[wid] = s1; aldst[wid] = s2; }
}

// ---- edge pass A: logits + segment max (atomicMax on uint keys) ------------

__global__ __launch_bounds__(256) void k_edge_logits_max(
    const int* __restrict__ srcv, const int* __restrict__ dstv,
    const float* __restrict__ ea, const float* __restrict__ Msm,
    const float* __restrict__ easum, const float* __restrict__ alsrc,
    const float* __restrict__ aldst, float* __restrict__ logits,
    unsigned* __restrict__ mkey, int heads) {
    int e = blockIdx.x * blockDim.x + threadIdx.x;
    if (e >= ETOT) return;
    int s, d;
    float feat[FE];
    if (e < EE) {
        s = srcv[e]; d = dstv[e];
        #pragma unroll
        for (int f = 0; f < FE; ++f) feat[f] = ea[(size_t)e * FE + f];
    } else {
        s = d = e - EE;
        const float invE = 1.0f / (float)EE;
        #pragma unroll
        for (int f = 0; f < FE; ++f) feat[f] = easum[f] * invE;
    }
    for (int hh = 0; hh < heads; ++hh) {
        float ale = 0.0f;
        #pragma unroll
        for (int f = 0; f < FE; ++f) ale += feat[f] * Msm[f * heads + hh];
        float lg = alsrc[s * heads + hh] + aldst[d * heads + hh] + ale;
        lg = (lg > 0.0f) ? lg : NEG_SLOPE * lg;
        logits[(size_t)e * heads + hh] = lg;
        atomicMax(&mkey[d * heads + hh], fkey(lg));
    }
}

// ---- edge pass B: p = exp(logit - max), segment sum ------------------------

__global__ __launch_bounds__(256) void k_edge_exp_sum(
    const int* __restrict__ dstv, float* __restrict__ logits,
    const unsigned* __restrict__ mkey, float* __restrict__ ssum, int heads) {
    int e = blockIdx.x * blockDim.x + threadIdx.x;
    if (e >= ETOT) return;
    int d = (e < EE) ? dstv[e] : (e - EE);
    for (int hh = 0; hh < heads; ++hh) {
        float m = funkey(mkey[d * heads + hh]);
        float p = __expf(logits[(size_t)e * heads + hh] - m);
        logits[(size_t)e * heads + hh] = p;
        atomicAdd(&ssum[d * heads + hh], p);
    }
}

// ---- edge pass B2: alpha = p / (sum + eps) ---------------------------------

__global__ __launch_bounds__(256) void k_alpha(
    const int* __restrict__ dstv, float* __restrict__ p,
    const float* __restrict__ ssum, int heads) {
    int e = blockIdx.x * blockDim.x + threadIdx.x;
    if (e >= ETOT) return;
    int d = (e < EE) ? dstv[e] : (e - EE);
    for (int hh = 0; hh < heads; ++hh)
        p[(size_t)e * heads + hh] /= (ssum[d * heads + hh] + 1e-16f);
}

// ---- edge pass C: out[dst] += h[src] * alpha (one block per edge) ----------
// blockDim = heads*C/4; one b128 load + 4 f32 atomics per thread.

__global__ void k_edge_scatter(
    const int* __restrict__ srcv, const int* __restrict__ dstv,
    const float* __restrict__ alpha, const float* __restrict__ h,
    float* __restrict__ acc, int heads, int C) {
    int e = blockIdx.x;
    int s, d;
    if (e < EE) { s = srcv[e]; d = dstv[e]; } else { s = d = e - EE; }
    int HC = heads * C;
    int i  = threadIdx.x << 2;                   // 4 consecutive channels
    int hh = i / C;
    float a = alpha[(size_t)e * heads + hh];
    const float4 hv = *(const float4*)(h + (size_t)s * HC + i);
    float* dp = acc + (size_t)d * HC + i;
    atomicAdd(dp + 0, hv.x * a);
    atomicAdd(dp + 1, hv.y * a);
    atomicAdd(dp + 2, hv.z * a);
    atomicAdd(dp + 3, hv.w * a);
}

// ---- layer norm (+bias, optional relu), one block per row ------------------

__global__ __launch_bounds__(256) void k_layernorm(
    const float* __restrict__ in, const float* __restrict__ bias,
    const float* __restrict__ gamma, const float* __restrict__ beta,
    float* __restrict__ out, int D, int doRelu) {
    __shared__ float red[256];
    int row = blockIdx.x, t = threadIdx.x;
    const float* r = in + (size_t)row * D;
    float s = 0.0f;
    for (int i = t; i < D; i += 256) s += r[i] + bias[i];
    red[t] = s; __syncthreads();
    for (int o = 128; o > 0; o >>= 1) { if (t < o) red[t] += red[t + o]; __syncthreads(); }
    float mu = red[0] / (float)D; __syncthreads();
    float v = 0.0f;
    for (int i = t; i < D; i += 256) { float x = r[i] + bias[i] - mu; v += x * x; }
    red[t] = v; __syncthreads();
    for (int o = 128; o > 0; o >>= 1) { if (t < o) red[t] += red[t + o]; __syncthreads(); }
    float rstd = rsqrtf(red[0] / (float)D + 1e-5f);
    for (int i = t; i < D; i += 256) {
        float x = (r[i] + bias[i] - mu) * rstd * gamma[i] + beta[i];
        out[(size_t)row * D + i] = doRelu ? fmaxf(x, 0.0f) : x;
    }
}

// ---------------------------------------------------------------------------

extern "C" void kernel_launch(void* const* d_in, const int* in_sizes, int n_in,
                              void* d_out, int out_size, void* d_ws, size_t ws_size,
                              hipStream_t stream) {
    (void)in_sizes; (void)n_in; (void)out_size; (void)ws_size;

    const float* x      = (const float*)d_in[0];
    const int*   eidx   = (const int*)  d_in[1];     // [2, E] int32
    const float* ea     = (const float*)d_in[2];
    const float* w1     = (const float*)d_in[3];
    const float* we1    = (const float*)d_in[4];
    const float* asrc1  = (const float*)d_in[5];
    const float* adst1  = (const float*)d_in[6];
    const float* aedge1 = (const float*)d_in[7];
    const float* b1     = (const float*)d_in[8];
    const float* g1     = (const float*)d_in[9];
    const float* beta1  = (const float*)d_in[10];
    const float* w2     = (const float*)d_in[11];
    const float* we2    = (const float*)d_in[12];
    const float* asrc2  = (const float*)d_in[13];
    const float* adst2  = (const float*)d_in[14];
    const float* aedge2 = (const float*)d_in[15];
    const float* b2     = (const float*)d_in[16];
    const float* g2     = (const float*)d_in[17];
    const float* beta2  = (const float*)d_in[18];
    float* out = (float*)d_out;

    const int* srcv = eidx;
    const int* dstv = eidx + EE;

    const int KP1 = 64;                      // FIN=55 padded to 64
    const int KP2 = HH * HIDC;               // 1024, already 32-aligned

    // ---- workspace layout (float-sized slots) ----
    float* W = (float*)d_ws;
    size_t o = 0;
    float* h1    = W + o; o += (size_t)NN * HH * HIDC;   // layer1 feats; reused as h2 after LN
    float* acc1  = W + o; o += (size_t)NN * HH * HIDC;   // layer1 accumulator; reused below
    float* z     = acc1;                                 // layer2 node feats (acc1 dead after LN)
    float* acc2  = acc1 + (size_t)NN * OUTC;             // layer2 accumulator
    float* logit = W + o; o += (size_t)ETOT * HH;        // logits/p/alpha; reused for layer2
    float* alsrc1v = W + o; o += (size_t)NN * HH;
    float* aldst1v = W + o; o += (size_t)NN * HH;
    float* m1v   = W + o; o += (size_t)NN * HH;          // uint keys (zero = -inf key)
    float* s1v   = W + o; o += (size_t)NN * HH;
    float* alsrc2v = W + o; o += (size_t)NN;
    float* aldst2v = W + o; o += (size_t)NN;
    float* m2v   = W + o; o += (size_t)NN;
    float* s2v   = W + o; o += (size_t)NN;
    float* M1    = W + o; o += FE * HH;
    float* M2    = W + o; o += FE + 2;
    float* easum = W + o; o += FE + 2;
    // bf16 GEMM operands (each uint = 2 bf16)
    unsigned* xbf  = (unsigned*)(W + o); o += (size_t)NN * KP1 / 2;
    unsigned* w1t  = (unsigned*)(W + o); o += (size_t)(HH * HIDC) * KP1 / 2;
    unsigned* h2bf = (unsigned*)(W + o); o += (size_t)NN * KP2 / 2;
    unsigned* w2t  = (unsigned*)(W + o); o += (size_t)OUTC * KP2 / 2;

    const int ZB = 2048, T = 256;
    auto cdiv = [](long a, long b) { return (int)((a + b - 1) / b); };

    // ---- init accumulators (every call: graph replay does not re-zero ws) ----
    k_zero<<<ZB, T, 0, stream>>>(acc1, (size_t)NN * HH * HIDC);
    k_zero<<<64,  T, 0, stream>>>(m1v,  (size_t)NN * HH);        // key 0 == -inf
    k_zero<<<64,  T, 0, stream>>>(s1v,  (size_t)NN * HH);
    k_zero<<<1,   64, 0, stream>>>(easum, FE);

    // ---- tiny precomputations ----
    k_ea_colsum<<<512, T, 0, stream>>>(ea, easum);
    k_make_M<<<1, 32, 0, stream>>>(we1, aedge1, we2, aedge2, M1, M2);

    // ---- pack layer1 GEMM operands to bf16 (A row-major padded, B transposed)
    k_cvt_pad<<<cdiv((long)NN * KP1 / 2, T), T, 0, stream>>>(x, xbf, NN, FIN, KP1);
    k_cvt_T_pad<<<cdiv((long)(HH * HIDC) * KP1 / 2, T), T, 0, stream>>>(
        w1, w1t, FIN, HH * HIDC, KP1);

    // ---- layer 1 ----
    // h1 = x @ W1   (M=20000, Nc=1024, Kp=64) via WMMA bf16
    {
        int tiles = (NN / 16) * ((HH * HIDC) / 64);
        k_gemm_wmma_bf16<<<cdiv(tiles, 4), 128, 0, stream>>>(
            (const __bf16*)xbf, (const __bf16*)w1t, h1, NN, HH * HIDC, KP1, HH * HIDC);
    }
    k_node_logits<<<cdiv((long)NN * HH * 32, T), T, 0, stream>>>(
        h1, asrc1, adst1, alsrc1v, aldst1v, NN, HH, HIDC);
    k_edge_logits_max<<<cdiv(ETOT, T), T, 0, stream>>>(
        srcv, dstv, ea, M1, easum, alsrc1v, aldst1v, logit, (unsigned*)m1v, HH);
    k_edge_exp_sum<<<cdiv(ETOT, T), T, 0, stream>>>(
        dstv, logit, (const unsigned*)m1v, s1v, HH);
    k_alpha<<<cdiv(ETOT, T), T, 0, stream>>>(dstv, logit, s1v, HH);
    k_edge_scatter<<<ETOT, (HH * HIDC) / 4, 0, stream>>>(
        srcv, dstv, logit, h1, acc1, HH, HIDC);
    // h2 = relu(LN(acc1 + b1)) -> written over h1 (h1 dead after scatter)
    k_layernorm<<<NN, T, 0, stream>>>(acc1, b1, g1, beta1, h1, HH * HIDC, 1);

    // ---- layer 2 ----
    k_zero<<<ZB, T, 0, stream>>>(acc2, (size_t)NN * OUTC);
    k_zero<<<64,  T, 0, stream>>>(m2v, (size_t)NN);
    k_zero<<<64,  T, 0, stream>>>(s2v, (size_t)NN);
    // pack layer2 GEMM operands
    k_cvt_pad<<<cdiv((long)NN * KP2 / 2, T), T, 0, stream>>>(h1, h2bf, NN, KP2, KP2);
    k_cvt_T_pad<<<cdiv((long)OUTC * KP2 / 2, T), T, 0, stream>>>(w2, w2t, KP2, OUTC, KP2);
    // z = h2 @ W2  (M=20000, Nc=256, Kp=1024) via WMMA bf16
    {
        int tiles = (NN / 16) * (OUTC / 64);
        k_gemm_wmma_bf16<<<cdiv(tiles, 4), 128, 0, stream>>>(
            (const __bf16*)h2bf, (const __bf16*)w2t, z, NN, OUTC, KP2, OUTC);
    }
    k_node_logits<<<cdiv((long)NN * 32, T), T, 0, stream>>>(
        z, asrc2, adst2, alsrc2v, aldst2v, NN, 1, OUTC);
    k_edge_logits_max<<<cdiv(ETOT, T), T, 0, stream>>>(
        srcv, dstv, ea, M2, easum, alsrc2v, aldst2v, logit, (unsigned*)m2v, 1);
    k_edge_exp_sum<<<cdiv(ETOT, T), T, 0, stream>>>(
        dstv, logit, (const unsigned*)m2v, s2v, 1);
    k_alpha<<<cdiv(ETOT, T), T, 0, stream>>>(dstv, logit, s2v, 1);
    k_edge_scatter<<<ETOT, OUTC / 4, 0, stream>>>(
        srcv, dstv, logit, z, acc2, 1, OUTC);
    // final: LN(acc2 + b2) -> d_out
    k_layernorm<<<NN, T, 0, stream>>>(acc2, b2, g2, beta2, out, OUTC, 0);
}
